// Memory_15118284882400
// MI455X (gfx1250) — matrix operand order repaired and checked by reading, not compiled, not run on hardware
//
#include <hip/hip_runtime.h>
#include <hip/hip_bf16.h>
#include <math.h>

typedef __bf16 bf16_t;
typedef bf16_t v16bf __attribute__((ext_vector_type(16)));
typedef bf16_t v8bf  __attribute__((ext_vector_type(8)));
typedef float  v8f   __attribute__((ext_vector_type(8)));

#define N_ROWS 65536   // B*H*W
#define CDIM   512
#define MDIM   512

// CDNA5 async global->LDS copies (ASYNCcnt path), if the toolchain declares them.
#if defined(__has_builtin)
#  if __has_builtin(__builtin_amdgcn_global_load_async_to_lds_b128) && \
      __has_builtin(__builtin_amdgcn_s_wait_asynccnt)
#    define USE_ASYNC_LDS 1
#  endif
#endif

#ifdef USE_ASYNC_LDS
// From the toolchain diagnostic: param 0 is an int4-vector pointer in the global
// (__device__ = addrspace(1)) address space; param 1 is the LDS-side pointer.
typedef int i32x4 __attribute__((ext_vector_type(4)));
typedef __attribute__((address_space(1))) i32x4 gi32x4;
typedef __attribute__((address_space(3))) i32x4 li32x4;
#endif

// ---------------------------------------------------------------- helpers
__device__ inline float blockReduceSum256(float v, float* buf, int t) {
  buf[t] = v; __syncthreads();
  #pragma unroll
  for (int s = 128; s > 0; s >>= 1) {
    if (t < s) buf[t] += buf[t + s];
    __syncthreads();
  }
  float r = buf[0]; __syncthreads();
  return r;
}

__device__ inline void atomicMaxFloat(float* addr, float v) {
  if (!(v < 0.0f)) atomicMax((int*)addr, __float_as_int(v));
  else             atomicMin((unsigned int*)addr, __float_as_uint(v));
}

__device__ inline void tile_barrier() {
#ifdef USE_ASYNC_LDS
  __builtin_amdgcn_s_wait_asynccnt(0);
#endif
  __syncthreads();
}

// ---------------------------------------------------------------- init
__global__ void init_kernel(float* colmax, float* colsum, float* qu,
                            float* sep, float* cmp) {
  int idx = blockIdx.x * 256 + threadIdx.x;
  if (idx < MDIM * CDIM) qu[idx] = 0.0f;
  if (idx < MDIM) { colmax[idx] = -3.0e38f; colsum[idx] = 0.0f; }
  if (idx == 0) { *sep = 0.0f; *cmp = 0.0f; }
}

// ---------------------------------------------------------------- keys -> bf16 (direct + transposed)
__global__ void keys2bf_kernel(const float* __restrict__ keys,
                               bf16_t* __restrict__ kbf, bf16_t* __restrict__ kbfT) {
  int idx = blockIdx.x * 256 + threadIdx.x;
  if (idx < MDIM * CDIM) {
    int m = idx >> 9, c = idx & 511;
    bf16_t v = (bf16_t)keys[idx];
    kbf[idx] = v;                    // [m][c]
    kbfT[(size_t)c * MDIM + m] = v;  // [c][m]
  }
}

// ---------------------------------------------------------------- normalize query
__global__ void normalize_kernel(const float* __restrict__ q,
                                 float* __restrict__ uq, bf16_t* __restrict__ qbf) {
  int w = threadIdx.x;
  int h = blockIdx.x * 4 + threadIdx.y;
  int b = blockIdx.y;
  size_t base = (((size_t)b * 512) * 64 + h) * 64 + w;   // q[b,0,h,w]
  float ss = 0.0f;
  for (int c = 0; c < CDIM; ++c) {
    float v = q[base + (size_t)c * 4096];
    ss += v * v;
  }
  float inv = 1.0f / fmaxf(sqrtf(ss), 1e-12f);
  size_t n  = ((size_t)(b * 64 + h)) * 64 + w;
  size_t ub = (((size_t)b * 1024) * 64 + h) * 64 + w;    // updated_query has 1024 ch
  for (int c = 0; c < CDIM; ++c) {
    float v = q[base + (size_t)c * 4096] * inv;
    uq[ub + (size_t)c * 4096] = v;
    qbf[n * CDIM + c] = (bf16_t)v;
  }
}

// ---------------------------------------------------------------- WMMA bf16 GEMM
// C[p, q] = sum_k A[p,k] * Bsrc[k, q]   (Bsrc row-major K x 512, ld = 512)
// Block tile 128x128, K-step 32, 8 waves (2 along P x 4 along Q), wave tile 64x32.
// Double-buffered LDS tiles; next K-tile staged with GLOBAL_LOAD_ASYNC_TO_LDS_B128
// (ASYNCcnt) while the current tile's 8 WMMAs execute.
//   As[.][row][k] row stride 40 elems (80 B):  A frag = two 16B runs -> ds_load_b128
//   Bs[.][k][q]   row stride 136 elems (272 B): B frag = 16 scalar 2B reads, one per k row (column gather)
__global__ __launch_bounds__(256) void wmma_gemm_bf16(
    const bf16_t* __restrict__ A, const bf16_t* __restrict__ Bsrc,
    float* __restrict__ C) {
  __shared__ __align__(16) bf16_t As[2][128][40];
  __shared__ __align__(16) bf16_t Bs[2][32][136];
  int tid  = threadIdx.x;
  int lane = tid & 31, wv = tid >> 5;
  int wq = wv & 3, wp = wv >> 2;
  int p0 = blockIdx.y * 128;
  int q0 = blockIdx.x * 128;
  int hi = lane >> 4, lo = lane & 15;

  v8f acc[4][2];
  #pragma unroll
  for (int i = 0; i < 4; ++i)
    #pragma unroll
    for (int j = 0; j < 2; ++j)
      #pragma unroll
      for (int e = 0; e < 8; ++e) acc[i][j][e] = 0.0f;

  auto stage = [&](int buf, int k0) {
#ifdef USE_ASYNC_LDS
    #pragma unroll
    for (int c = tid; c < 512; c += 256) {
      int r = c >> 2, kk0 = (c & 3) * 8;
      __builtin_amdgcn_global_load_async_to_lds_b128(
          (gi32x4*)(const void*)&A[(size_t)(p0 + r) * CDIM + k0 + kk0],
          (li32x4*)(void*)&As[buf][r][kk0], 0, 0);
    }
    #pragma unroll
    for (int c = tid; c < 512; c += 256) {
      int k = c >> 4, qq0 = (c & 15) * 8;
      __builtin_amdgcn_global_load_async_to_lds_b128(
          (gi32x4*)(const void*)&Bsrc[(size_t)(k0 + k) * CDIM + q0 + qq0],
          (li32x4*)(void*)&Bs[buf][k][qq0], 0, 0);
    }
#else
    #pragma unroll
    for (int c = tid; c < 512; c += 256) {
      int r = c >> 2, kk0 = (c & 3) * 8;
      *(v8bf*)&As[buf][r][kk0] = *(const v8bf*)&A[(size_t)(p0 + r) * CDIM + k0 + kk0];
    }
    #pragma unroll
    for (int c = tid; c < 512; c += 256) {
      int k = c >> 4, qq0 = (c & 15) * 8;
      *(v8bf*)&Bs[buf][k][qq0] = *(const v8bf*)&Bsrc[(size_t)(k0 + k) * CDIM + q0 + qq0];
    }
#endif
  };

  stage(0, 0);
  tile_barrier();

  int buf = 0;
  for (int k0 = 0; k0 < CDIM; k0 += 32) {
    if (k0 + 32 < CDIM) stage(buf ^ 1, k0 + 32);  // overlap next tile's copies with WMMA

    v16bf a[4], b[2];
    #pragma unroll
    for (int j = 0; j < 2; ++j) {
      int col0 = wq * 32 + 16 * j;
      // CDNA5 B layout: lane lo holds col N, K = {8*hi..8*hi+7, 8*hi+16..8*hi+23}
      #pragma unroll
      for (int s = 0; s < 16; ++s)
        b[j][s] = Bs[buf][(s < 8 ? 8 * hi + s : 16 + 8 * hi + (s - 8))][col0 + lo];
    }
    #pragma unroll
    for (int i = 0; i < 4; ++i) {
      int row = wp * 64 + 16 * i + lo;
      // CDNA5 16-bit A layout: lane lo holds row, K = {8*hi..8*hi+7, 8*hi+16..8*hi+23}
      v8bf alo = *(const v8bf*)&As[buf][row][8 * hi];
      v8bf ahi = *(const v8bf*)&As[buf][row][8 * hi + 16];
      a[i] = __builtin_shufflevector(alo, ahi, 0, 1, 2, 3, 4, 5, 6, 7,
                                               8, 9, 10, 11, 12, 13, 14, 15);
    }
    #pragma unroll
    for (int i = 0; i < 4; ++i)
      #pragma unroll
      for (int j = 0; j < 2; ++j)
        acc[i][j] = __builtin_amdgcn_wmma_f32_16x16x32_bf16(
            false, a[i], false, b[j], (short)0, acc[i][j], false, false);

    tile_barrier();   // next tile's async copies complete; safe to flip
    buf ^= 1;
  }

  // epilogue: C/D layout — lane&15 = N col, VGPR v + 8*(lane>>4) = M row
  #pragma unroll
  for (int i = 0; i < 4; ++i)
    #pragma unroll
    for (int j = 0; j < 2; ++j)
      #pragma unroll
      for (int v = 0; v < 8; ++v) {
        int row = p0 + wp * 64 + 16 * i + v + 8 * hi;
        int col = q0 + wq * 32 + 16 * j + lo;
        C[(size_t)row * MDIM + col] = acc[i][j][v];
      }
}

// ---------------------------------------------------------------- row softmax + top2 + losses
__global__ __launch_bounds__(256) void rowsoftmax_kernel(
    const float* __restrict__ score, const bf16_t* __restrict__ qbf,
    const float* __restrict__ keys,
    float* __restrict__ sm, bf16_t* __restrict__ smbf,
    int* __restrict__ g, float* __restrict__ sctop,
    float* __restrict__ sep_out, float* __restrict__ cmp_out) {
  __shared__ float rv[256];
  __shared__ int   ri[256];
  int n = blockIdx.x, t = threadIdx.x;
  size_t rb = (size_t)n * MDIM;
  float v0 = score[rb + t], v1 = score[rb + t + 256];

  // ---- top1 (argmax, lowest index on tie)
  float mv; int mi;
  if (v0 >= v1) { mv = v0; mi = t; } else { mv = v1; mi = t + 256; }
  rv[t] = mv; ri[t] = mi; __syncthreads();
  #pragma unroll
  for (int s = 128; s > 0; s >>= 1) {
    if (t < s) {
      if (rv[t + s] > rv[t] || (rv[t + s] == rv[t] && ri[t + s] < ri[t])) {
        rv[t] = rv[t + s]; ri[t] = ri[t + s];
      }
    }
    __syncthreads();
  }
  float M1 = rv[0]; int I1 = ri[0]; __syncthreads();

  // ---- top2 (exclude I1)
  mv = -3.0e38f; mi = 0;
  if (t != I1) { mv = v0; mi = t; }
  if ((t + 256) != I1 && (v1 > mv || (v1 == mv && (t + 256) < mi))) { mv = v1; mi = t + 256; }
  rv[t] = mv; ri[t] = mi; __syncthreads();
  #pragma unroll
  for (int s = 128; s > 0; s >>= 1) {
    if (t < s) {
      if (rv[t + s] > rv[t] || (rv[t + s] == rv[t] && ri[t + s] < ri[t])) {
        rv[t] = rv[t + s]; ri[t] = ri[t + s];
      }
    }
    __syncthreads();
  }
  int I2 = ri[0]; __syncthreads();

  // ---- softmax over row
  float e0 = __expf(v0 - M1), e1 = __expf(v1 - M1);
  float S = blockReduceSum256(e0 + e1, rv, t);
  float inv = 1.0f / S;
  float s0 = e0 * inv, s1 = e1 * inv;
  sm[rb + t] = s0; sm[rb + t + 256] = s1;
  smbf[rb + t] = (bf16_t)s0; smbf[rb + t + 256] = (bf16_t)s1;

  // ---- losses (pos = keys[I1], neg = keys[I2])
  float ca = 0.0f, dp = 0.0f, dnm = 0.0f;
  for (int c = t; c < CDIM; c += 256) {
    float qv = (float)qbf[rb + c];
    float pv = keys[(size_t)I1 * CDIM + c];
    float nv = keys[(size_t)I2 * CDIM + c];
    float d = qv - pv;
    ca += d * d;
    float a = d + 1e-6f;        dp  += a * a;
    float bb = qv - nv + 1e-6f; dnm += bb * bb;
  }
  float CA = blockReduceSum256(ca, rv, t);
  float DP = blockReduceSum256(dp, rv, t);
  float DN = blockReduceSum256(dnm, rv, t);
  if (t == 0) {
    g[n] = I1; sctop[n] = M1;
    atomicAdd(cmp_out, CA * (1.0f / ((float)N_ROWS * (float)CDIM)));
    float sv = sqrtf(DP) - sqrtf(DN) + 1.0f;
    sv = sv > 0.0f ? sv : 0.0f;
    atomicAdd(sep_out, sv * (1.0f / (float)N_ROWS));
  }
}

// ---------------------------------------------------------------- column max / sum / s_query
__global__ void colmax_kernel(const float* __restrict__ score, float* __restrict__ colmax) {
  int tx = threadIdx.x, ty = threadIdx.y;          // (32,8)
  int m = blockIdx.x * 32 + tx;
  int n0 = blockIdx.y * 2048;
  float mv = -3.0e38f;
  for (int i = ty; i < 2048; i += 8)
    mv = fmaxf(mv, score[(size_t)(n0 + i) * MDIM + m]);
  __shared__ float buf[8][32];
  buf[ty][tx] = mv; __syncthreads();
  if (ty == 0) {
    float r = buf[0][tx];
    #pragma unroll
    for (int j = 1; j < 8; ++j) r = fmaxf(r, buf[j][tx]);
    atomicMaxFloat(&colmax[m], r);
  }
}

__global__ void colsum_kernel(const float* __restrict__ score,
                              const float* __restrict__ colmax, float* __restrict__ colsum) {
  int tx = threadIdx.x, ty = threadIdx.y;
  int m = blockIdx.x * 32 + tx;
  int n0 = blockIdx.y * 2048;
  float cmv = colmax[m];
  float s = 0.0f;
  for (int i = ty; i < 2048; i += 8)
    s += __expf(score[(size_t)(n0 + i) * MDIM + m] - cmv);
  __shared__ float buf[8][32];
  buf[ty][tx] = s; __syncthreads();
  if (ty == 0) {
    float r = 0.0f;
    #pragma unroll
    for (int j = 0; j < 8; ++j) r += buf[j][tx];
    atomicAdd(&colsum[m], r);
  }
}

__global__ void squery_kernel(const float* __restrict__ score,
                              const float* __restrict__ colmax,
                              const float* __restrict__ colsum, float* __restrict__ sq) {
  size_t idx = (size_t)blockIdx.x * 256 + threadIdx.x;
  int m = (int)(idx & 511);
  sq[idx] = __expf(score[idx] - colmax[m]) / colsum[m];
}

// ---------------------------------------------------------------- NHWC -> NCHW scatter (both outs)
__global__ void scatter_cm_kernel(const float* __restrict__ cm,
                                  float* __restrict__ uq, float* __restrict__ uo) {
  __shared__ float tile[32][33];
  int tx = threadIdx.x, ty = threadIdx.y;   // (32,8)
  int cb = blockIdx.x & 15, wb = blockIdx.x >> 4;
  int h = blockIdx.y, b = blockIdx.z;
  int c0 = cb * 32, w0 = wb * 32;
  size_t n0 = ((size_t)(b * 64 + h)) * 64 + w0;
  #pragma unroll
  for (int ii = 0; ii < 4; ++ii) {
    int w = ty + 8 * ii;
    tile[w][tx] = cm[(n0 + w) * CDIM + c0 + tx];
  }
  __syncthreads();
  #pragma unroll
  for (int ii = 0; ii < 4; ++ii) {
    int c = ty + 8 * ii;
    float v = tile[tx][c];
    int cg = c0 + c;
    uq[(((size_t)b * 1024 + 512 + cg) * 64 + h) * 64 + w0 + tx] = v;
    uo[(((size_t)b * 512 + cg) * 64 + h) * 64 + w0 + tx] = v;
  }
}

// ---------------------------------------------------------------- segment-sum memory update
__global__ void segsum_kernel(const int* __restrict__ g, const float* __restrict__ sctop,
                              const float* __restrict__ colmax,
                              const bf16_t* __restrict__ qbf, float* __restrict__ qu) {
  int n = blockIdx.x, t = threadIdx.x;
  int gi = g[n];
  // coeff = s_query[n, gi] / colmax(s_query)[gi] == exp(score[n,gi] - colmax_score[gi])
  float coeff = __expf(sctop[n] - colmax[gi]);
  size_t rb = (size_t)n * CDIM, ob = (size_t)gi * CDIM;
  for (int c = t; c < CDIM; c += 256)
    atomicAdd(&qu[ob + c], coeff * (float)qbf[rb + c]);
}

// ---------------------------------------------------------------- updated_memory = l2norm(qu+keys)
__global__ void updmem_kernel(const float* __restrict__ qu, const float* __restrict__ keys,
                              float* __restrict__ um) {
  int m = blockIdx.x, t = threadIdx.x;
  size_t rb = (size_t)m * CDIM;
  float a0 = qu[rb + t] + keys[rb + t];
  float a1 = qu[rb + t + 256] + keys[rb + t + 256];
  __shared__ float rv[256];
  float S = blockReduceSum256(a0 * a0 + a1 * a1, rv, t);
  float inv = 1.0f / fmaxf(sqrtf(S), 1e-12f);
  um[rb + t] = a0 * inv; um[rb + t + 256] = a1 * inv;
}

// ---------------------------------------------------------------- launch
extern "C" void kernel_launch(void* const* d_in, const int* in_sizes, int n_in,
                              void* d_out, int out_size, void* d_ws, size_t ws_size,
                              hipStream_t stream) {
  (void)in_sizes; (void)n_in; (void)out_size; (void)ws_size;
  const float* q    = (const float*)d_in[0];   // (16,512,64,64)
  const float* keys = (const float*)d_in[1];   // (512,512)

  float* out = (float*)d_out;
  float* uq  = out;                           // updated_query  (16,1024,64,64)
  float* uo  = out + 67108864;                // updated_orig   (16,512,64,64)
  float* um  = out + 100663296;               // updated_memory (512,512)
  float* sq  = out + 100925440;               // s_query  (65536,512)
  float* sm  = out + 134479872;               // s_memory (65536,512)
  float* sep = out + 168034304;               // separateness_loss
  float* cmp = out + 168034305;               // compactness_loss

  unsigned char* w = (unsigned char*)d_ws;
  bf16_t* qbf   = (bf16_t*)w;                   w += (size_t)N_ROWS * CDIM * 2;   // 64 MB
  bf16_t* kbf   = (bf16_t*)w;                   w += (size_t)MDIM * CDIM * 2;
  bf16_t* kbfT  = (bf16_t*)w;                   w += (size_t)MDIM * CDIM * 2;
  float*  score = (float*)w;                    w += (size_t)N_ROWS * MDIM * 4;   // 128 MB (reused as concat_memory)
  bf16_t* smbf  = (bf16_t*)w;                   w += (size_t)N_ROWS * MDIM * 2;   // 64 MB
  float*  colmax= (float*)w;                    w += MDIM * 4;
  float*  colsum= (float*)w;                    w += MDIM * 4;
  float*  sctop = (float*)w;                    w += N_ROWS * 4;
  int*    g     = (int*)w;                      w += N_ROWS * 4;
  float*  qu    = (float*)w;                    w += (size_t)MDIM * CDIM * 4;
  float*  cmbuf = score;                        // alias: score dead after squery

  init_kernel<<<1024, 256, 0, stream>>>(colmax, colsum, qu, sep, cmp);
  keys2bf_kernel<<<1024, 256, 0, stream>>>(keys, kbf, kbfT);
  normalize_kernel<<<dim3(16, 16), dim3(64, 4), 0, stream>>>(q, uq, qbf);

  // score = qr @ keys^T : B[k=c][q=m] = keys^T (c-major)
  wmma_gemm_bf16<<<dim3(MDIM / 128, N_ROWS / 128), 256, 0, stream>>>(qbf, kbfT, score);

  rowsoftmax_kernel<<<N_ROWS, 256, 0, stream>>>(score, qbf, keys, sm, smbf, g, sctop, sep, cmp);

  colmax_kernel<<<dim3(16, 32), dim3(32, 8), 0, stream>>>(score, colmax);
  colsum_kernel<<<dim3(16, 32), dim3(32, 8), 0, stream>>>(score, colmax, colsum);
  squery_kernel<<<131072, 256, 0, stream>>>(score, colmax, colsum, sq);

  // concat_memory = s_memory @ keys : B[k=m][q=c] = keys (m-major)
  wmma_gemm_bf16<<<dim3(CDIM / 128, N_ROWS / 128), 256, 0, stream>>>(smbf, kbf, cmbuf);

  scatter_cm_kernel<<<dim3(32, 64, 16), dim3(32, 8), 0, stream>>>(cmbuf, uq, uo);
  segsum_kernel<<<N_ROWS, 256, 0, stream>>>(g, sctop, colmax, qbf, qu);
  updmem_kernel<<<MDIM, 256, 0, stream>>>(qu, keys, um);
}
